// GCA_32839319945339
// MI455X (gfx1250) — compile-verified, matrix operand order checked by reading, not run on hardware
//
#include <hip/hip_runtime.h>
#include <hip/hip_bf16.h>

// ---------------- problem constants ----------------
#define NNODES 40000
#define NC     128
#define NH     8
#define HD     16
#define NE     640000
#define NFF    512
#define EH     (NE*NH)        // 5,120,000
#define EH_BLOCKS (EH/256)    // 20,000
#define ROW_BLOCKS (NNODES/16)

typedef __bf16 bf16_t;
typedef bf16_t v16bf __attribute__((ext_vector_type(16)));
typedef float  v8f   __attribute__((ext_vector_type(8)));

// ---------------- helpers ----------------
__device__ __forceinline__ unsigned short f2bf(float f) {
  unsigned u = __float_as_uint(f);
  u += 0x7FFFu + ((u >> 16) & 1u);          // round-to-nearest-even
  return (unsigned short)(u >> 16);
}

// pair->K mapping for 16-bit A/B fragments (ISA 7.12.2):
// VGPR p holds K pair starting at: p<4 -> 8*hi+2p ; p>=4 -> 16+8*hi+2(p-4)
__device__ __forceinline__ int kpair(int p, int hi) {
  return (p < 4) ? (8*hi + 2*p) : (16 + 8*hi + 2*(p - 4));
}

// Build 16x32 bf16 A fragment from LDS (row-major ushort, row stride SA, SA even)
__device__ __forceinline__ v16bf load_a_frag(const unsigned short* ldsA, int SA,
                                             int kb, int lane) {
  int hi = lane >> 4, row = lane & 15;
  union { v16bf v; unsigned u[8]; } f;
  const unsigned short* rp = ldsA + row * SA + kb * 32;
#pragma unroll
  for (int p = 0; p < 8; ++p)
    f.u[p] = *(const unsigned*)(rp + kpair(p, hi));   // 4B aligned (even k)
  return f.v;
}

// B fragments are pre-packed contiguously: ((nt*KB + kb)*32 + lane)*16 ushorts
__device__ __forceinline__ v16bf load_b_frag(const unsigned short* pack, int nt,
                                             int kb, int KB, int lane) {
  return *(const v16bf*)(pack + (((size_t)(nt * KB + kb)) * 32 + lane) * 16);
}

__device__ __forceinline__ v8f wmma_bf16(v16bf a, v16bf b, v8f c) {
  return __builtin_amdgcn_wmma_f32_16x16x32_bf16(false, a, false, b,
                                                 (short)0, c, false, false);
}

// ---------------- kernel: pack fp32 W[K][N] into WMMA B-fragment bf16 layout ----
__global__ void k_pack(const float* __restrict__ W, unsigned short* __restrict__ P,
                       int K, int Nn) {
  int t = blockIdx.x * blockDim.x + threadIdx.x;
  int total = (Nn / 16) * (K / 32) * 32;
  if (t >= total) return;
  int lane = t & 31;
  int frag = t >> 5;
  int KB = K / 32;
  int kb = frag % KB;
  int nt = frag / KB;
  int hi = lane >> 4, idx = lane & 15;
  unsigned short* dst = P + (size_t)t * 16;
#pragma unroll
  for (int j = 0; j < 16; ++j) {
    int kk = kb * 32 + ((j < 8) ? (8*hi + j) : (16 + 8*hi + (j - 8)));
    dst[j] = f2bf(W[(size_t)kk * Nn + nt * 16 + idx]);
  }
}

// ---------------- kernel: LN1 + fused QKV GEMM ----------------
__global__ void k_ln1_qkv(const float* __restrict__ x,
                          const float* __restrict__ g, const float* __restrict__ b,
                          const unsigned short* __restrict__ Wq_p, const float* __restrict__ bq,
                          const unsigned short* __restrict__ Wk_p, const float* __restrict__ bk,
                          const unsigned short* __restrict__ Wv_p, const float* __restrict__ bv,
                          float* __restrict__ oq, float* __restrict__ ok,
                          float* __restrict__ ov) {
  __shared__ float tile[16 * 128];
  __shared__ float mean[16], rstd[16];
  __shared__ __align__(16) unsigned short ldsA[16 * 130];
  int rb  = blockIdx.x * 16;
  int tid = threadIdx.x;

  for (int i = tid; i < 2048; i += 256) tile[i] = x[(size_t)rb * 128 + i];
  __syncthreads();
  if (tid < 16) {
    const float* r = tile + tid * 128;
    float s = 0.f, s2 = 0.f;
    for (int i = 0; i < 128; ++i) { float v = r[i]; s += v; s2 += v * v; }
    float m = s * (1.f / 128.f);
    float var = s2 * (1.f / 128.f) - m * m;
    mean[tid] = m; rstd[tid] = rsqrtf(var + 1e-5f);
  }
  __syncthreads();
  for (int i = tid; i < 2048; i += 256) {
    int r = i >> 7, c = i & 127;
    float xn = (tile[i] - mean[r]) * rstd[r] * g[c] + b[c];
    ldsA[r * 130 + c] = f2bf(xn);
  }
  __syncthreads();

  int wave = tid >> 5, lane = tid & 31;
  int hi = lane >> 4, idx = lane & 15;
  const unsigned short* Wp[3]  = { Wq_p, Wk_p, Wv_p };
  const float*          bia[3] = { bq, bk, bv };
  float*                out[3] = { oq, ok, ov };
#pragma unroll
  for (int m = 0; m < 3; ++m) {
    v8f acc = {0.f,0.f,0.f,0.f,0.f,0.f,0.f,0.f};
#pragma unroll
    for (int kb = 0; kb < 4; ++kb) {
      v16bf a = load_a_frag(ldsA, 130, kb, lane);
      v16bf bb = load_b_frag(Wp[m], wave, kb, 4, lane);
      acc = wmma_bf16(a, bb, acc);
    }
    int n = wave * 16 + idx;
    float bv_ = bia[m][n];
#pragma unroll
    for (int r = 0; r < 8; ++r) {
      int row = rb + 8 * hi + r;
      out[m][(size_t)row * 128 + n] = acc[r] + bv_;
    }
  }
}

// ---------------- kernel: per-edge scores + per-block head max ----------------
__global__ void k_scores(const float* __restrict__ q, const float* __restrict__ k,
                         const int* __restrict__ ei, float* __restrict__ scores,
                         float* __restrict__ pmax) {
  int t = blockIdx.x * 256 + threadIdx.x;
  int h = threadIdx.x & 7;
  float s = -3.402823466e38f;
  if (t < EH) {
    int e = t >> 3;
    int src = ei[e], dst = ei[NE + e];
    const float4* qp = (const float4*)(q + (size_t)dst * 128 + h * 16);
    const float4* kp = (const float4*)(k + (size_t)src * 128 + h * 16);
    float acc = 0.f;
#pragma unroll
    for (int i = 0; i < 4; ++i) {
      float4 a = qp[i], b = kp[i];
      acc += a.x*b.x + a.y*b.y + a.z*b.z + a.w*b.w;
    }
    s = acc * 0.25f;           // 1/sqrt(16)
    scores[t] = s;
  }
  __shared__ float sm[256];
  sm[threadIdx.x] = s;
  __syncthreads();
  if (threadIdx.x < 8) {
    float m = sm[threadIdx.x];
    for (int i = 1; i < 32; ++i) m = fmaxf(m, sm[i * 8 + threadIdx.x]);
    pmax[blockIdx.x * 8 + threadIdx.x] = m;
  }
}

__global__ void k_reduce_max(const float* __restrict__ p, int nb,
                             float* __restrict__ hmax) {
  int h = threadIdx.x & 7, g = threadIdx.x >> 3;
  float m = -3.402823466e38f;
  for (int b = g; b < nb; b += 32) m = fmaxf(m, p[b * 8 + h]);
  __shared__ float sm[256];
  sm[threadIdx.x] = m;
  __syncthreads();
  if (threadIdx.x < 8) {
    float mm = sm[threadIdx.x];
    for (int i = 1; i < 32; ++i) mm = fmaxf(mm, sm[i * 8 + threadIdx.x]);
    hmax[threadIdx.x] = mm;
  }
}

__global__ void k_expsum(float* __restrict__ scores, const float* __restrict__ hmax,
                         float* __restrict__ psum) {
  int t = blockIdx.x * 256 + threadIdx.x;
  int h = threadIdx.x & 7;
  float sv = 0.f;
  if (t < EH) {
    float ex = __expf(scores[t] - hmax[h]);
    scores[t] = ex;
    sv = ex;
  }
  __shared__ float sm[256];
  sm[threadIdx.x] = sv;
  __syncthreads();
  if (threadIdx.x < 8) {
    float s = sm[threadIdx.x];
    for (int i = 1; i < 32; ++i) s += sm[i * 8 + threadIdx.x];
    psum[blockIdx.x * 8 + threadIdx.x] = s;
  }
}

__global__ void k_reduce_sum(const float* __restrict__ p, int nb,
                             float* __restrict__ hsuminv) {
  int h = threadIdx.x & 7, g = threadIdx.x >> 3;
  float s = 0.f;
  for (int b = g; b < nb; b += 32) s += p[b * 8 + h];
  __shared__ float sm[256];
  sm[threadIdx.x] = s;
  __syncthreads();
  if (threadIdx.x < 8) {
    float ss = sm[threadIdx.x];
    for (int i = 1; i < 32; ++i) ss += sm[i * 8 + threadIdx.x];
    hsuminv[threadIdx.x] = 1.f / ss;
  }
}

// ---------------- kernel: weighted scatter-aggregate ----------------
__global__ void k_aggregate(const float* __restrict__ scores, const float* __restrict__ v,
                            const int* __restrict__ ei, const float* __restrict__ hsuminv,
                            float* __restrict__ agg) {
  int t = blockIdx.x * 256 + threadIdx.x;
  if (t >= EH) return;
  int e = t >> 3, h = t & 7;
  int src = ei[e], dst = ei[NE + e];
  float w = scores[t] * hsuminv[h];
  const float* vp = v + (size_t)src * 128 + h * 16;
  float*       ap = agg + (size_t)dst * 128 + h * 16;
#pragma unroll
  for (int d = 0; d < 16; ++d) atomicAdd(ap + d, vp[d] * w);
}

// ---------------- kernel: out-proj + residual + LN2 ----------------
__global__ void k_attnout_ln2(const float* __restrict__ agg,
                              const unsigned short* __restrict__ Wo_p,
                              const float* __restrict__ bo,
                              const float* __restrict__ x,
                              const float* __restrict__ g2, const float* __restrict__ b2,
                              float* __restrict__ x1, unsigned short* __restrict__ xn2) {
  __shared__ float tile[16 * 128];
  __shared__ float mean[16], rstd[16];
  __shared__ __align__(16) unsigned short ldsA[16 * 130];
  int rb = blockIdx.x * 16;
  int tid = threadIdx.x;

  for (int i = tid; i < 2048; i += 256) {
    int r = i >> 7, c = i & 127;
    ldsA[r * 130 + c] = f2bf(agg[(size_t)rb * 128 + i]);
  }
  __syncthreads();

  int wave = tid >> 5, lane = tid & 31;
  int hi = lane >> 4, idx = lane & 15;
  {
    v8f acc = {0.f,0.f,0.f,0.f,0.f,0.f,0.f,0.f};
#pragma unroll
    for (int kb = 0; kb < 4; ++kb) {
      v16bf a = load_a_frag(ldsA, 130, kb, lane);
      v16bf bb = load_b_frag(Wo_p, wave, kb, 4, lane);
      acc = wmma_bf16(a, bb, acc);
    }
    int n = wave * 16 + idx;
    float bo_ = bo[n];
#pragma unroll
    for (int r = 0; r < 8; ++r) {
      int row = rb + 8 * hi + r;
      float v1 = x[(size_t)row * 128 + n] + acc[r] + bo_;
      x1[(size_t)row * 128 + n] = v1;
      tile[(8 * hi + r) * 128 + n] = v1;
    }
  }
  __syncthreads();
  if (tid < 16) {
    const float* r = tile + tid * 128;
    float s = 0.f, s2 = 0.f;
    for (int i = 0; i < 128; ++i) { float v = r[i]; s += v; s2 += v * v; }
    float m = s * (1.f / 128.f);
    float var = s2 * (1.f / 128.f) - m * m;
    mean[tid] = m; rstd[tid] = rsqrtf(var + 1e-5f);
  }
  __syncthreads();
  for (int i = tid; i < 2048; i += 256) {
    int r = i >> 7, c = i & 127;
    float xn = (tile[i] - mean[r]) * rstd[r] * g2[c] + b2[c];
    xn2[(size_t)rb * 128 + i] = f2bf(xn);
  }
}

// ---------------- kernel: FFN1 (128 -> 512, relu, bf16 hidden) ----------------
__global__ void k_ffn1(const unsigned short* __restrict__ xn2,
                       const unsigned short* __restrict__ W1_p,
                       const float* __restrict__ b1,
                       unsigned short* __restrict__ hid) {
  __shared__ __align__(16) unsigned short ldsA[16 * 130];
  int rb = blockIdx.x * 16;
  int tid = threadIdx.x;
  for (int i = tid; i < 2048; i += 256) {
    int r = i >> 7, c = i & 127;
    ldsA[r * 130 + c] = xn2[(size_t)(rb + r) * 128 + c];
  }
  __syncthreads();
  int wave = tid >> 5, lane = tid & 31;
  int hi = lane >> 4, idx = lane & 15;
#pragma unroll
  for (int t = 0; t < 4; ++t) {
    int nt = wave * 4 + t;          // 32 column tiles of 16
    v8f acc = {0.f,0.f,0.f,0.f,0.f,0.f,0.f,0.f};
#pragma unroll
    for (int kb = 0; kb < 4; ++kb) {
      v16bf a = load_a_frag(ldsA, 130, kb, lane);
      v16bf bb = load_b_frag(W1_p, nt, kb, 4, lane);
      acc = wmma_bf16(a, bb, acc);
    }
    int n = nt * 16 + idx;
    float b1_ = b1[n];
#pragma unroll
    for (int r = 0; r < 8; ++r) {
      int row = rb + 8 * hi + r;
      float h = acc[r] + b1_;
      hid[(size_t)row * 512 + n] = f2bf(h > 0.f ? h : 0.f);
    }
  }
}

// ---------------- kernel: FFN2 (512 -> 128) + residual2 ----------------
__global__ void k_ffn2(const unsigned short* __restrict__ hid,
                       const unsigned short* __restrict__ W2_p,
                       const float* __restrict__ b2,
                       const float* __restrict__ x1,
                       float* __restrict__ out) {
  __shared__ __align__(16) unsigned short ldsA[16 * 514];
  int rb = blockIdx.x * 16;
  int tid = threadIdx.x;
  for (int i = tid; i < 8192; i += 256) {
    int r = i >> 9, c = i & 511;
    ldsA[r * 514 + c] = hid[(size_t)(rb + r) * 512 + c];
  }
  __syncthreads();
  int wave = tid >> 5, lane = tid & 31;
  int hi = lane >> 4, idx = lane & 15;
  v8f acc = {0.f,0.f,0.f,0.f,0.f,0.f,0.f,0.f};
#pragma unroll
  for (int kb = 0; kb < 16; ++kb) {
    v16bf a = load_a_frag(ldsA, 514, kb, lane);
    v16bf bb = load_b_frag(W2_p, wave, kb, 16, lane);
    acc = wmma_bf16(a, bb, acc);
  }
  int n = wave * 16 + idx;
  float b2_ = b2[n];
#pragma unroll
  for (int r = 0; r < 8; ++r) {
    int row = rb + 8 * hi + r;
    out[(size_t)row * 128 + n] = x1[(size_t)row * 128 + n] + acc[r] + b2_;
  }
}

// ---------------- host launch ----------------
extern "C" void kernel_launch(void* const* d_in, const int* in_sizes, int n_in,
                              void* d_out, int out_size, void* d_ws, size_t ws_size,
                              hipStream_t stream) {
  const float* x    = (const float*)d_in[0];
  const int*   ei   = (const int*)d_in[1];
  const float* Wq   = (const float*)d_in[2];
  const float* bq   = (const float*)d_in[3];
  const float* Wk   = (const float*)d_in[4];
  const float* bk   = (const float*)d_in[5];
  const float* Wv   = (const float*)d_in[6];
  const float* bv   = (const float*)d_in[7];
  const float* Wo   = (const float*)d_in[8];
  const float* bo   = (const float*)d_in[9];
  const float* l1g  = (const float*)d_in[10];
  const float* l1b  = (const float*)d_in[11];
  const float* l2g  = (const float*)d_in[12];
  const float* l2b  = (const float*)d_in[13];
  const float* W1   = (const float*)d_in[14];
  const float* b1   = (const float*)d_in[15];
  const float* W2   = (const float*)d_in[16];
  const float* b2   = (const float*)d_in[17];
  float* out = (float*)d_out;

  char* ws = (char*)d_ws;
  const size_t SZ_NC  = (size_t)NNODES * NC * sizeof(float);   // 20,480,000
  const size_t SZ_EH  = (size_t)EH * sizeof(float);            // 20,480,000
  float* q      = (float*)(ws + 0);
  float* kk     = (float*)(ws + SZ_NC);
  float* vv     = (float*)(ws + 2 * SZ_NC);
  float* scores = (float*)(ws + 3 * SZ_NC);
  float* agg    = (float*)(ws + 3 * SZ_NC + SZ_EH);
  float* x1     = (float*)(ws + 4 * SZ_NC + SZ_EH);
  unsigned short* xn2 = (unsigned short*)(ws + 5 * SZ_NC + SZ_EH);
  size_t off = 5 * SZ_NC + SZ_EH + (size_t)NNODES * NC * 2;     // 133,120,000
  float* pmax = (float*)(ws + off);              off += (size_t)EH_BLOCKS * 8 * 4;
  float* psum = (float*)(ws + off);              off += (size_t)EH_BLOCKS * 8 * 4;
  float* hmax = (float*)(ws + off);              off += 256;
  float* hsi  = (float*)(ws + off);              off += 256;
  unsigned short* Wq_p = (unsigned short*)(ws + off); off += 128 * 128 * 2;
  unsigned short* Wk_p = (unsigned short*)(ws + off); off += 128 * 128 * 2;
  unsigned short* Wv_p = (unsigned short*)(ws + off); off += 128 * 128 * 2;
  unsigned short* Wo_p = (unsigned short*)(ws + off); off += 128 * 128 * 2;
  unsigned short* W1_p = (unsigned short*)(ws + off); off += 128 * 512 * 2;
  unsigned short* W2_p = (unsigned short*)(ws + off); off += 512 * 128 * 2;
  // hidden reuses the (dead after k_scores) q+k region: exactly 40,960,000 B
  unsigned short* hid = (unsigned short*)(ws + 0);

  // pack weights into WMMA B-fragment layout
  k_pack<<<4,  256, 0, stream>>>(Wq, Wq_p, 128, 128);
  k_pack<<<4,  256, 0, stream>>>(Wk, Wk_p, 128, 128);
  k_pack<<<4,  256, 0, stream>>>(Wv, Wv_p, 128, 128);
  k_pack<<<4,  256, 0, stream>>>(Wo, Wo_p, 128, 128);
  k_pack<<<16, 256, 0, stream>>>(W1, W1_p, 128, 512);
  k_pack<<<16, 256, 0, stream>>>(W2, W2_p, 512, 128);

  // LN1 + QKV
  k_ln1_qkv<<<ROW_BLOCKS, 256, 0, stream>>>(x, l1g, l1b, Wq_p, bq, Wk_p, bk,
                                            Wv_p, bv, q, kk, vv);
  // attention scores + global softmax (per head, over all edges)
  k_scores<<<EH_BLOCKS, 256, 0, stream>>>(q, kk, ei, scores, pmax);
  k_reduce_max<<<1, 256, 0, stream>>>(pmax, EH_BLOCKS, hmax);
  k_expsum<<<EH_BLOCKS, 256, 0, stream>>>(scores, hmax, psum);
  k_reduce_sum<<<1, 256, 0, stream>>>(psum, EH_BLOCKS, hsi);
  // scatter aggregate
  hipMemsetAsync(agg, 0, SZ_NC, stream);
  k_aggregate<<<EH_BLOCKS, 256, 0, stream>>>(scores, vv, ei, hsi, agg);
  // out-proj + residual + LN2
  k_attnout_ln2<<<ROW_BLOCKS, 256, 0, stream>>>(agg, Wo_p, bo, x, l2g, l2b, x1, xn2);
  // FFN
  k_ffn1<<<ROW_BLOCKS, 256, 0, stream>>>(xn2, W1_p, b1, hid);
  k_ffn2<<<ROW_BLOCKS, 256, 0, stream>>>(hid, W2_p, b2, x1, out);
}
